// Change_Guide_Mul_Window_attention_67345087201627
// MI455X (gfx1250) — matrix operand order repaired and checked by reading, compiled, not run on hardware
//
#include <hip/hip_runtime.h>
#include <hip/hip_bf16.h>
#include <math.h>

typedef __attribute__((ext_vector_type(16))) _Float16 v16h;
typedef __attribute__((ext_vector_type(8)))  _Float16 v8h;
typedef __attribute__((ext_vector_type(8)))  float    v8f;

#define SCALE_ATTN 0.17677669529663687f  // 32^-0.5

__device__ inline v16h cat8(v8h a, v8h b) {
  return __builtin_shufflevector(a, b, 0,1,2,3,4,5,6,7,8,9,10,11,12,13,14,15);
}

// ---- A-tile (16x32 f16, M=rows of activations, K=channels) contiguous k order (same as ldB):
// lane<16: M=lane, K=0..15 contiguous; lane>=16: M=lane-16, K=16..31
__device__ inline v16h ldA(const _Float16* A, int row0, int ldk, int k0, int lane) {
  int r = lane & 15, h = lane >> 4;
  const _Float16* p = A + (size_t)(row0 + r) * ldk + k0 + h * 16;
  v8h a0 = *(const v8h*)p;
  v8h a1 = *(const v8h*)(p + 8);
  return cat8(a0, a1);
}

// ---- B-tile (32x16 f16). B[k,n] = W[n, k] (weights row-major [out,in]):
// lane<16: n=lane, k=0..15 contiguous; lane>=16: n=lane-16, k=16..31
__device__ inline v16h ldB(const _Float16* W, int col0, int ldk, int k0, int lane) {
  int n = lane & 15, h = lane >> 4;
  const _Float16* p = W + (size_t)(col0 + n) * ldk + k0 + h * 16;
  return *(const v16h*)p;
}

// ================= prep kernels =================
__global__ void cvt_weights_k(const float* w_qkv, const float* w_v, const float* w_proj,
                              const float* w_temp, _Float16* wtemp16, _Float16* wqkv16,
                              _Float16* wv16, _Float16* wproj16) {
  int i = blockIdx.x * blockDim.x + threadIdx.x;      // 393216 total
  if (i < 131072)       wtemp16[i]          = (_Float16)w_temp[i];
  else if (i < 262144)  wqkv16[i - 131072]  = (_Float16)w_qkv[i - 131072];
  else if (i < 327680)  wv16[i - 262144]    = (_Float16)w_v[i - 262144];
  else                  wproj16[i - 327680] = (_Float16)w_proj[i - 327680];
}

__global__ void pack_x_k(const float* x, _Float16* xp) {
  int i = blockIdx.x * blockDim.x + threadIdx.x;      // 8*256*4096
  int p = i & 4095;
  int c = (i >> 12) & 255;
  int b = i >> 20;
  xp[((size_t)(b * 4096 + p)) * 256 + c] = (_Float16)x[i];
}

// bilinear 16x16 -> 64x64 (half-pixel centers, clamped == jax renormalized), write [b][p][c] f16
__global__ void resize_t_k(const float* temp, _Float16* tres) {
  int i = blockIdx.x * blockDim.x + threadIdx.x;      // 8*4096*512 = 2^24
  int c = i & 511;
  int p = (i >> 9) & 4095;
  int b = i >> 21;
  int y = p >> 6, xx = p & 63;
  float fy = (y + 0.5f) * 0.25f - 0.5f;
  float fx = (xx + 0.5f) * 0.25f - 0.5f;
  int y0 = (int)floorf(fy); float wy = fy - (float)y0;
  int x0 = (int)floorf(fx); float wx = fx - (float)x0;
  int y0c = min(max(y0, 0), 15), y1c = min(y0 + 1, 15);
  int x0c = min(max(x0, 0), 15), x1c = min(x0 + 1, 15);
  const float* tb = temp + ((size_t)b * 512 + c) * 256;
  float v00 = tb[y0c * 16 + x0c], v01 = tb[y0c * 16 + x1c];
  float v10 = tb[y1c * 16 + x0c], v11 = tb[y1c * 16 + x1c];
  float v = (1.f - wy) * ((1.f - wx) * v00 + wx * v01)
          +        wy  * ((1.f - wx) * v10 + wx * v11);
  tres[i] = (_Float16)v;
}

// ================= generic GEMM: C[M,N] = A[M,K] * W[N,K]^T (+bias), f16 out =================
__global__ void __launch_bounds__(256) gemm16_k(const _Float16* __restrict__ A,
                                                const _Float16* __restrict__ W,
                                                _Float16* __restrict__ C,
                                                int N, int K, const float* bias) {
  int wave = (blockIdx.x * blockDim.x + threadIdx.x) >> 5;
  int lane = threadIdx.x & 31;
  int nGroups = N >> 6;
  int mt = wave / nGroups, ng = wave - mt * nGroups;
  int row0 = mt * 16, col0 = ng * 64;
  v8f acc0 = {}, acc1 = {}, acc2 = {}, acc3 = {};
  for (int k = 0; k < K; k += 32) {
    v16h a  = ldA(A, row0, K, k, lane);
    __builtin_prefetch(A + (size_t)(row0 + (lane & 15)) * K + k + 32, 0, 1);
    v16h b0 = ldB(W, col0,      K, k, lane);
    v16h b1 = ldB(W, col0 + 16, K, k, lane);
    v16h b2 = ldB(W, col0 + 32, K, k, lane);
    v16h b3 = ldB(W, col0 + 48, K, k, lane);
    acc0 = __builtin_amdgcn_wmma_f32_16x16x32_f16(false, a, false, b0, (short)0, acc0, false, false);
    acc1 = __builtin_amdgcn_wmma_f32_16x16x32_f16(false, a, false, b1, (short)0, acc1, false, false);
    acc2 = __builtin_amdgcn_wmma_f32_16x16x32_f16(false, a, false, b2, (short)0, acc2, false, false);
    acc3 = __builtin_amdgcn_wmma_f32_16x16x32_f16(false, a, false, b3, (short)0, acc3, false, false);
  }
  int colL = lane & 15, rowH = (lane >> 4) * 8;
  v8f accs[4] = {acc0, acc1, acc2, acc3};
#pragma unroll
  for (int t = 0; t < 4; ++t) {
#pragma unroll
    for (int j = 0; j < 8; ++j) {
      int row = row0 + rowH + j;
      int col = col0 + t * 16 + colL;
      float v = accs[t][j] + (bias ? bias[col] : 0.f);
      C[(size_t)row * N + col] = (_Float16)v;
    }
  }
}

// projection GEMM: out[b,o,p] = attn[row,:]*w_proj[o,:] + b_proj[o] + x[b,o,p]  (f32, NCHW)
__global__ void __launch_bounds__(256) gemm_proj_k(const _Float16* __restrict__ A,
                                                   const _Float16* __restrict__ W,
                                                   const float* __restrict__ bias,
                                                   const float* __restrict__ x,
                                                   float* __restrict__ out) {
  const int N = 256, K = 256;
  int wave = (blockIdx.x * blockDim.x + threadIdx.x) >> 5;
  int lane = threadIdx.x & 31;
  int mt = wave >> 2, ng = wave & 3;
  int row0 = mt * 16, col0 = ng * 64;
  v8f acc0 = {}, acc1 = {}, acc2 = {}, acc3 = {};
  for (int k = 0; k < K; k += 32) {
    v16h a  = ldA(A, row0, K, k, lane);
    v16h b0 = ldB(W, col0,      K, k, lane);
    v16h b1 = ldB(W, col0 + 16, K, k, lane);
    v16h b2 = ldB(W, col0 + 32, K, k, lane);
    v16h b3 = ldB(W, col0 + 48, K, k, lane);
    acc0 = __builtin_amdgcn_wmma_f32_16x16x32_f16(false, a, false, b0, (short)0, acc0, false, false);
    acc1 = __builtin_amdgcn_wmma_f32_16x16x32_f16(false, a, false, b1, (short)0, acc1, false, false);
    acc2 = __builtin_amdgcn_wmma_f32_16x16x32_f16(false, a, false, b2, (short)0, acc2, false, false);
    acc3 = __builtin_amdgcn_wmma_f32_16x16x32_f16(false, a, false, b3, (short)0, acc3, false, false);
  }
  int colL = lane & 15, rowH = (lane >> 4) * 8;
  v8f accs[4] = {acc0, acc1, acc2, acc3};
#pragma unroll
  for (int t = 0; t < 4; ++t) {
#pragma unroll
    for (int j = 0; j < 8; ++j) {
      int row = row0 + rowH + j;           // = b*4096 + p
      int col = col0 + t * 16 + colL;      // output channel
      int b = row >> 12, p = row & 4095;
      size_t idx = ((size_t)b * N + col) * 4096 + p;
      out[idx] = accs[t][j] + bias[col] + x[idx];
    }
  }
}

// ================= small-window attention (heads 0,1; s=4,16), scalar =================
__global__ void attn_small_k(const _Float16* __restrict__ q16,
                             const _Float16* __restrict__ kv16,
                             _Float16* __restrict__ attn16) {
  int t = blockIdx.x * blockDim.x + threadIdx.x;   // 2*8*4096
  int head = t >> 15;
  int b = (t >> 12) & 7;
  int p = t & 4095;
  int ws = (head == 0) ? 2 : 4;
  int s = ws * ws;
  int y = p >> 6, x = p & 63;
  int gy = (y / ws) * ws, gx = (x / ws) * ws;
  int c0 = head * 32;
  const _Float16* qp = q16 + ((size_t)(b * 4096 + p)) * 256 + c0;
  float qv[32];
#pragma unroll
  for (int d = 0; d < 32; ++d) qv[d] = (float)qp[d];
  float sc[16];
  float mx = -1e30f;
  for (int j = 0; j < s; ++j) {
    int pj = (gy + j / ws) * 64 + gx + j % ws;
    const _Float16* kp = kv16 + ((size_t)(b * 4096 + pj)) * 512 + c0;
    float dot = 0.f;
#pragma unroll
    for (int d = 0; d < 32; ++d) dot += qv[d] * (float)kp[d];
    sc[j] = dot * SCALE_ATTN;
    mx = fmaxf(mx, sc[j]);
  }
  float sum = 0.f;
  float acc[32];
#pragma unroll
  for (int d = 0; d < 32; ++d) acc[d] = 0.f;
  for (int j = 0; j < s; ++j) {
    int pj = (gy + j / ws) * 64 + gx + j % ws;
    float e = __expf(sc[j] - mx);
    sum += e;
    const _Float16* vp = kv16 + ((size_t)(b * 4096 + pj)) * 512 + 256 + c0;
#pragma unroll
    for (int d = 0; d < 32; ++d) acc[d] += e * (float)vp[d];
  }
  _Float16* op = attn16 + ((size_t)(b * 4096 + p)) * 256 + c0;
  float inv = 1.f / sum;
#pragma unroll
  for (int d = 0; d < 32; ++d) op[d] = (_Float16)(acc[d] * inv);
}

// ================= flash WMMA attention: heads 2..4 (window 8/16/32) + heads 6..8 (row, s=64) ====
// one wave per 16-query block; online softmax; P bounced through LDS; V staged into LDS via
// GLOBAL_LOAD_ASYNC_TO_LDS_B128 (ASYNCcnt DMA, no VGPR round-trip).
__global__ void __launch_bounds__(128) attn_flash_k(const _Float16* __restrict__ q16,
                                                    const _Float16* __restrict__ kv16,
                                                    _Float16* __restrict__ attn16) {
  __shared__ _Float16 smem[4 * 1536];
  int waveInBlk = threadIdx.x >> 5;
  int lane = threadIdx.x & 31;
  _Float16* pbuf = smem + waveInBlk * 1536;   // [16][32] P staging
  _Float16* vbuf = pbuf + 512;                // [32(k)][32(ch)] V tile, natural layout

  int Wv = blockIdx.x * 4 + waveInBlk;        // 0..12287
  int seg = Wv >> 11;
  int r = Wv & 2047;
  int head, s, b, qb, gy = 0, gx = 0, ws = 1, lws = 0, rowy = 0;
  bool isRow = false;
  if (seg == 0)      { head = 2; ws = 8;  lws = 3; s = 64;   b = r >> 8; int g = (r >> 2) & 63; gy = (g >> 3) * 8;  gx = (g & 7) * 8;  qb = r & 3;  }
  else if (seg == 1) { head = 3; ws = 16; lws = 4; s = 256;  b = r >> 8; int g = (r >> 4) & 15; gy = (g >> 2) * 16; gx = (g & 3) * 16; qb = r & 15; }
  else if (seg == 2) { head = 4; ws = 32; lws = 5; s = 1024; b = r >> 8; int g = (r >> 6) & 3;  gy = (g >> 1) * 32; gx = (g & 1) * 32; qb = r & 63; }
  else               { head = seg + 3; s = 64; b = r >> 8; rowy = (r >> 2) & 63; qb = r & 3; isRow = true; }

  auto PIX = [&](int i) -> int {
    if (isRow) return rowy * 64 + i;
    return (gy + (i >> lws)) * 64 + gx + (i & (ws - 1));
  };

  int rloc = lane & 15, half = lane >> 4;
  int c0 = head * 32;

  // Q A-tile (16 rows x K=32 channels)
  int iq = qb * 16 + rloc;
  const _Float16* qp = q16 + ((size_t)(b * 4096 + PIX(iq))) * 256 + c0 + half * 16;
  v16h qa = cat8(*(const v8h*)qp, *(const v8h*)(qp + 8));

  // per-lane async-DMA addressing for V staging: 4 x B128, lane l covers row l>>2, 16B chunk l&3
  int vrow = lane >> 2;                     // 0..7
  int vseg = lane & 3;                      // 16B chunk within 64B row
  unsigned vbase = (unsigned)(size_t)vbuf + (unsigned)(vrow * 64 + vseg * 16);

  float m[8], l[8];
#pragma unroll
  for (int j = 0; j < 8; ++j) { m[j] = -1e30f; l[j] = 0.f; }
  v8f oa = {}, ob = {};

  for (int j0 = 0; j0 < s; j0 += 32) {
    // make sure prior-iteration LDS reads of vbuf retired before DMA overwrites it
    __asm__ volatile("s_wait_dscnt 0" ::: "memory");
    // stage V chunk (32 seq rows x 32 ch, natural [k][ch] layout) via async DMA to LDS
#pragma unroll
    for (int i = 0; i < 4; ++i) {
      int jv = j0 + i * 8 + vrow;
      const _Float16* gp = kv16 + ((size_t)(b * 4096 + PIX(jv))) * 512 + 256 + c0 + vseg * 8;
      unsigned lds = vbase + (unsigned)(i * 8 * 64);
      unsigned long long ga = (unsigned long long)(size_t)gp;
      __asm__ volatile("global_load_async_to_lds_b128 %0, %1, off"
                       :: "v"(lds), "v"(ga) : "memory");
    }

    // K B-tiles: B[k=d, n=j] = K[pix(j), d] (contiguous per lane)
    const _Float16* kp1 = kv16 + ((size_t)(b * 4096 + PIX(j0 + rloc)))      * 512 + c0 + half * 16;
    const _Float16* kp2 = kv16 + ((size_t)(b * 4096 + PIX(j0 + 16 + rloc))) * 512 + c0 + half * 16;
    v16h kb1 = *(const v16h*)kp1;
    v16h kb2 = *(const v16h*)kp2;
    v8f z = {};
    v8f sa = __builtin_amdgcn_wmma_f32_16x16x32_f16(false, qa, false, kb1, (short)0, z, false, false);
    v8f sb = __builtin_amdgcn_wmma_f32_16x16x32_f16(false, qa, false, kb2, (short)0, z, false, false);

    // online softmax per row (rows: half*8+j; 16 columns across lanes of this half)
#pragma unroll
    for (int j = 0; j < 8; ++j) {
      float s1 = sa[j] * SCALE_ATTN, s2 = sb[j] * SCALE_ATTN;
      float tmax = fmaxf(s1, s2);
      tmax = fmaxf(tmax, __shfl_xor(tmax, 1, 16));
      tmax = fmaxf(tmax, __shfl_xor(tmax, 2, 16));
      tmax = fmaxf(tmax, __shfl_xor(tmax, 4, 16));
      tmax = fmaxf(tmax, __shfl_xor(tmax, 8, 16));
      float nm = fmaxf(m[j], tmax);
      float corr = __expf(m[j] - nm);
      float p1 = __expf(s1 - nm), p2 = __expf(s2 - nm);
      float rs = p1 + p2;
      rs += __shfl_xor(rs, 1, 16);
      rs += __shfl_xor(rs, 2, 16);
      rs += __shfl_xor(rs, 4, 16);
      rs += __shfl_xor(rs, 8, 16);
      l[j] = l[j] * corr + rs;
      m[j] = nm;
      oa[j] *= corr;
      ob[j] *= corr;
      pbuf[(half * 8 + j) * 32 + rloc]      = (_Float16)p1;
      pbuf[(half * 8 + j) * 32 + 16 + rloc] = (_Float16)p2;
    }

    __asm__ volatile("" ::: "memory");
    __builtin_amdgcn_wave_barrier();
    __asm__ volatile("s_wait_dscnt 0" ::: "memory");
    __asm__ volatile("s_wait_asynccnt 0" ::: "memory");

    // P as A-tile (16x32)
    const _Float16* pp = pbuf + rloc * 32 + half * 16;
    v16h pA = cat8(*(const v8h*)pp, *(const v8h*)(pp + 8));

    // V B-tiles from vbuf[k][ch]: lane column n = ch, k strided by 32 halves
    int kb = half * 16;
    v16h vB1 = {}, vB2 = {};
#pragma unroll
    for (int t = 0; t < 16; ++t) {
      vB1[t] = vbuf[(kb + t) * 32 + rloc];
      vB2[t] = vbuf[(kb + t) * 32 + 16 + rloc];
    }

    oa = __builtin_amdgcn_wmma_f32_16x16x32_f16(false, pA, false, vB1, (short)0, oa, false, false);
    ob = __builtin_amdgcn_wmma_f32_16x16x32_f16(false, pA, false, vB2, (short)0, ob, false, false);

    __asm__ volatile("" ::: "memory");
    __builtin_amdgcn_wave_barrier();
  }

#pragma unroll
  for (int j = 0; j < 8; ++j) {
    float inv = 1.f / l[j];
    int i = qb * 16 + half * 8 + j;
    size_t po = ((size_t)(b * 4096 + PIX(i))) * 256 + c0;
    attn16[po + rloc]      = (_Float16)(oa[j] * inv);
    attn16[po + 16 + rloc] = (_Float16)(ob[j] * inv);
  }
}

// ================= launch =================
extern "C" void kernel_launch(void* const* d_in, const int* in_sizes, int n_in,
                              void* d_out, int out_size, void* d_ws, size_t ws_size,
                              hipStream_t stream) {
  (void)in_sizes; (void)n_in; (void)out_size; (void)ws_size;
  const float* x      = (const float*)d_in[0];
  const float* temp   = (const float*)d_in[1];
  const float* w_qkv  = (const float*)d_in[2];
  const float* w_v    = (const float*)d_in[3];
  const float* w_proj = (const float*)d_in[4];
  const float* b_proj = (const float*)d_in[5];
  const float* w_temp = (const float*)d_in[6];
  const float* b_temp = (const float*)d_in[7];
  float* out = (float*)d_out;

  char* ws = (char*)d_ws;
  _Float16* wtemp16 = (_Float16*)(ws);
  _Float16* wqkv16  = (_Float16*)(ws + 262144);
  _Float16* wv16    = (_Float16*)(ws + 524288);
  _Float16* wproj16 = (_Float16*)(ws + 655360);
  _Float16* tres  = (_Float16*)(ws + (1 << 20));       // 32768 x 512  (32 MB)
  _Float16* xp    = tres  + (size_t)32768 * 512;       // 32768 x 256  (16 MB)
  _Float16* tproj = xp    + (size_t)32768 * 256;       // 32768 x 256  (16 MB)
  _Float16* kv    = tproj + (size_t)32768 * 256;       // 32768 x 512  (32 MB)
  _Float16* q     = kv    + (size_t)32768 * 512;       // 32768 x 256  (16 MB)
  _Float16* attnb = q     + (size_t)32768 * 256;       // 32768 x 256  (16 MB)

  cvt_weights_k<<<1536, 256, 0, stream>>>(w_qkv, w_v, w_proj, w_temp,
                                          wtemp16, wqkv16, wv16, wproj16);
  resize_t_k<<<65536, 256, 0, stream>>>(temp, tres);
  pack_x_k<<<32768, 256, 0, stream>>>(x, xp);

  // t_proj = tres * w_temp^T + b_temp   (K=512, N=256)
  gemm16_k<<<1024, 256, 0, stream>>>(tres, wtemp16, tproj, 256, 512, b_temp);
  // kv = xp * w_qkv^T                   (K=256, N=512)
  gemm16_k<<<2048, 256, 0, stream>>>(xp, wqkv16, kv, 512, 256, nullptr);
  // q = tproj * w_v^T                   (K=256, N=256)
  gemm16_k<<<1024, 256, 0, stream>>>(tproj, wv16, q, 256, 256, nullptr);

  attn_small_k<<<256, 256, 0, stream>>>(q, kv, attnb);
  attn_flash_k<<<3072, 128, 0, stream>>>(q, kv, attnb);

  gemm_proj_k<<<1024, 256, 0, stream>>>(attnb, wproj16, b_proj, x, out);
}
